// DWaveNet_47966194762019
// MI455X (gfx1250) — matrix operand order repaired and checked by reading, not compiled, not run on hardware
//
#include <hip/hip_runtime.h>

typedef _Float16 v16h __attribute__((ext_vector_type(16)));
typedef _Float16 v8h  __attribute__((ext_vector_type(8)));
typedef float    v8f  __attribute__((ext_vector_type(8)));

#define T_FULL   16384
#define RCH      128
#define T2       10246
#define TPAD     10304        /* 161 * 64 */
#define LDIFF    3069
#define FC1C     2048
#define FC2C     256
#define NLAYERS  30

// ---------------- workspace layout (bytes) ----------------
#define OFF_HA   ((size_t)0)
#define SZ_H     ((size_t)4*RCH*T_FULL*4)            /* 33,554,432 */
#define OFF_HB   (OFF_HA + SZ_H)
#define OFF_SKIP (OFF_HB + SZ_H)
#define OFF_Y1   (OFF_SKIP + SZ_H)
#define SZ_Y1    ((size_t)4*FC1C*TPAD*2)             /* 168,820,736 */
#define OFF_WD   (OFF_Y1 + SZ_Y1)
#define SZ_WD    ((size_t)NLAYERS*128*384*2)
#define OFF_WS   (OFF_WD + SZ_WD)
#define SZ_WSR   ((size_t)NLAYERS*128*64*2)
#define OFF_WR   (OFF_WS + SZ_WSR)
#define OFF_WF1  (OFF_WR + SZ_WSR)
#define SZ_WF1   ((size_t)FC1C*384*2)
#define OFF_WF2  (OFF_WF1 + SZ_WF1)

// ---------------- helpers ----------------
__device__ __forceinline__ v8f wmma_f16(v16h a, v16h b, v8f c) {
  return __builtin_amdgcn_wmma_f32_16x16x32_f16(
      false, a, false, b, (short)0, c, false, false);
}

// B-fragment scatter index for a 16-wide N tile group; K within [0, 32*nsteps)
__device__ __forceinline__ int frag_idx(int K, int n, int nsteps) {
  int lane = (n & 15) | (((K >> 3) & 1) << 4);
  return (((n >> 4) * nsteps + (K >> 5)) * 32 + lane) * 16 + ((K & 7) | (((K >> 4) & 1) << 3));
}

// A-fragment load from f16 weights laid out [row][ldw] row-major.
// Matches 16-bit A 16x32 layout: lanes<16 K 0-7 & 16-23; lanes>=16 K 8-15 & 24-31.
__device__ __forceinline__ v16h load_afrag(const _Float16* __restrict__ W,
                                           int ldw, int row, int kstep, int lane) {
  int kbase = kstep * 32 + ((lane >= 16) ? 8 : 0);
  const _Float16* p = W + (size_t)row * ldw + kbase;
  v8h lo = *(const v8h*)(p);
  v8h hi = *(const v8h*)(p + 16);
  return __builtin_shufflevector(lo, hi, 0,1,2,3,4,5,6,7,8,9,10,11,12,13,14,15);
}

__device__ __forceinline__ v8f zero8() {
  v8f z = {0.f,0.f,0.f,0.f,0.f,0.f,0.f,0.f};
  return z;
}

// ---------------- weight prep: f32 -> f16 in [Cout][K=tap*Cin+ci] order ----
__global__ void prep_weights(const float* __restrict__ Wd,
                             const float* __restrict__ Ws,
                             const float* __restrict__ Wr,
                             const float* __restrict__ Wf1,
                             const float* __restrict__ Wf2,
                             _Float16* __restrict__ wd16,
                             _Float16* __restrict__ ws16,
                             _Float16* __restrict__ wr16,
                             _Float16* __restrict__ wf116,
                             _Float16* __restrict__ wf216) {
  const int Nd  = NLAYERS*128*128*3;
  const int Nsr = NLAYERS*128*64;
  const int Nf1 = FC1C*128*3;
  const int Nf2 = FC2C*2048*3;
  int i = blockIdx.x * 256 + threadIdx.x;
  if (i < Nd) {
    int k = i % 3, r = i / 3;
    int ci = r % 128; r /= 128;
    int co = r % 128; int l = r / 128;
    wd16[((size_t)l*128 + co)*384 + k*128 + ci] = (_Float16)Wd[i];
    return;
  }
  i -= Nd;
  if (i < Nsr) { ws16[i] = (_Float16)Ws[i]; return; }
  i -= Nsr;
  if (i < Nsr) { wr16[i] = (_Float16)Wr[i]; return; }
  i -= Nsr;
  if (i < Nf1) {
    int k = i % 3, r = i / 3;
    int ci = r % 128; int co = r / 128;
    wf116[(size_t)co*384 + k*128 + ci] = (_Float16)Wf1[i];
    return;
  }
  i -= Nf1;
  if (i < Nf2) {
    int k = i % 3, r = i / 3;
    int ci = r % 2048; int co = r / 2048;
    wf216[(size_t)co*6144 + k*2048 + ci] = (_Float16)Wf2[i];
  }
}

// ---------------- first conv: Cin=1, K=3, same pad -------------------------
__global__ void first_conv(const float* __restrict__ x,
                           const float* __restrict__ Wf,
                           float* __restrict__ h) {
  size_t i = (size_t)blockIdx.x * 256 + threadIdx.x;
  if (i >= (size_t)4*RCH*T_FULL) return;
  int t = (int)(i % T_FULL);
  size_t r = i / T_FULL;
  int co = (int)(r % RCH);
  int b  = (int)(r / RCH);
  const float* xb = x + (size_t)b * T_FULL;
  float acc = 0.f;
  #pragma unroll
  for (int k = 0; k < 3; ++k) {
    int tt = t + k - 1;
    if (tt >= 0 && tt < T_FULL) acc += Wf[co*3 + k] * xb[tt];
  }
  h[i] = acc;
}

__global__ void zero_buf(float* __restrict__ p, size_t n) {
  size_t i = (size_t)blockIdx.x * 256 + threadIdx.x;
  if (i < n) p[i] = 0.f;
}

// ---------------- fused WaveNet layer --------------------------------------
// grid: 4 * (T/128) = 512 blocks; 256 threads = 8 waves.
// Block tile: M=128 x N=128. Wave = 2 M-tiles x 4 N-tiles (8 wmma / B-frag pair).
// K=384 staged in two 192-deep LDS chunks (48KB each, reusing one buffer).
__global__ __launch_bounds__(256) void layer_kernel(
    const float* __restrict__ hin, float* __restrict__ hout,
    float* __restrict__ skips,
    const _Float16* __restrict__ Wd, const _Float16* __restrict__ Wsk,
    const _Float16* __restrict__ Wrs, int dil) {
  __shared__ __align__(32) unsigned char smem[50176];
  _Float16* bfrag = (_Float16*)smem;                 // 192K x 128n frags = 48KB
  _Float16* g16   = (_Float16*)smem;                 // 128 x 132 halfs = 33792B
  _Float16* zfrag = (_Float16*)(smem + 33792);       // 8192 halfs = 16KB

  int tid = threadIdx.x;
  int bx  = blockIdx.x;
  int b   = bx >> 7;                                 // T/128 = 128 tiles
  int t0  = (bx & 127) << 7;

  int w = tid >> 5, l = tid & 31;
  int col   = l & 15;
  int rsub  = (l >= 16) ? 8 : 0;
  int mt2   = (w & 3) * 2;                           // wave's first M-tile
  int nbase = (w >> 2) * 4;                          // wave's first N-tile

  // pull this layer's weight panel toward the caches (global_prefetch_b8)
  if (tid < 128) {
    __builtin_prefetch(Wd + (size_t)tid * 384, 0, 1);
    __builtin_prefetch(Wsk + (size_t)tid * 64, 0, 1);
    __builtin_prefetch(Wrs + (size_t)tid * 64, 0, 1);
  }

  v8f acc[2][4];
  #pragma unroll
  for (int m = 0; m < 2; ++m)
    #pragma unroll
    for (int nt = 0; nt < 4; ++nt) acc[m][nt] = zero8();

  // ---- GEMM1: g = Wdil * taps  (K = tap*128+ci, two 192-K chunks)
  for (int c = 0; c < 2; ++c) {
    for (int e = tid; e < 192*128; e += 256) {
      int Kl = e >> 7, n = e & 127;
      int K = c*192 + Kl;
      int k = K >> 7, ci = K & 127;
      int t = t0 + n + (k - 1) * dil;
      float v = 0.f;
      if (t >= 0 && t < T_FULL) v = hin[((size_t)b*RCH + ci)*T_FULL + t];
      bfrag[frag_idx(Kl, n, 6)] = (_Float16)v;
    }
    __syncthreads();
    #pragma unroll
    for (int s6 = 0; s6 < 6; ++s6) {
      int s = c*6 + s6;
      v16h a0 = load_afrag(Wd, 384, (mt2+0)*16 + col, s, l);
      v16h a1 = load_afrag(Wd, 384, (mt2+1)*16 + col, s, l);
      #pragma unroll
      for (int nt = 0; nt < 4; ++nt) {
        v16h bm = *(const v16h*)&bfrag[(((nbase+nt)*6 + s6)*32 + l)*16];
        acc[0][nt] = wmma_f16(a0, bm, acc[0][nt]);
        acc[1][nt] = wmma_f16(a1, bm, acc[1][nt]);
      }
    }
    __syncthreads();
  }

  // ---- exchange g through LDS (f16) for the cross-wave GLU
  #pragma unroll
  for (int m = 0; m < 2; ++m) {
    int rb = (mt2+m)*16 + rsub;
    #pragma unroll
    for (int nt = 0; nt < 4; ++nt) {
      int n = (nbase+nt)*16 + col;
      #pragma unroll
      for (int q = 0; q < 8; ++q)
        g16[(rb + q)*132 + n] = (_Float16)acc[m][nt][q];
    }
  }
  __syncthreads();

  // ---- GLU: z = tanh(a) * sigmoid(b), write as B-fragments (K=64 -> 2 steps)
  for (int e = tid; e < 64*128; e += 256) {
    int cch = e >> 7, n = e & 127;
    float av = (float)g16[cch*132 + n];
    float bv = (float)g16[(cch + 64)*132 + n];
    float z = tanhf(av) * (1.f / (1.f + expf(-bv)));
    zfrag[frag_idx(cch, n, 2)] = (_Float16)z;
  }
  __syncthreads();

  const float scale = 0.70710678118654752f;

  // ---- GEMM2a: skip = Wsk * z  (M=128, K=64), accumulate into skips buffer
  {
    v8f sk[2][4];
    #pragma unroll
    for (int m = 0; m < 2; ++m)
      #pragma unroll
      for (int nt = 0; nt < 4; ++nt) sk[m][nt] = zero8();
    #pragma unroll
    for (int s = 0; s < 2; ++s) {
      v16h a0 = load_afrag(Wsk, 64, (mt2+0)*16 + col, s, l);
      v16h a1 = load_afrag(Wsk, 64, (mt2+1)*16 + col, s, l);
      #pragma unroll
      for (int nt = 0; nt < 4; ++nt) {
        v16h bm = *(const v16h*)&zfrag[(((nbase+nt)*2 + s)*32 + l)*16];
        sk[0][nt] = wmma_f16(a0, bm, sk[0][nt]);
        sk[1][nt] = wmma_f16(a1, bm, sk[1][nt]);
      }
    }
    #pragma unroll
    for (int m = 0; m < 2; ++m)
      #pragma unroll
      for (int nt = 0; nt < 4; ++nt) {
        int t = t0 + (nbase+nt)*16 + col;
        #pragma unroll
        for (int q = 0; q < 8; ++q) {
          size_t idx = ((size_t)b*RCH + (mt2+m)*16 + rsub + q)*T_FULL + t;
          skips[idx] += sk[m][nt][q];
        }
      }
  }

  // ---- GEMM2b: res = Wrs * z; h' = (h + res) * sqrt(0.5)  (ping-pong)
  {
    v8f rs[2][4];
    #pragma unroll
    for (int m = 0; m < 2; ++m)
      #pragma unroll
      for (int nt = 0; nt < 4; ++nt) rs[m][nt] = zero8();
    #pragma unroll
    for (int s = 0; s < 2; ++s) {
      v16h a0 = load_afrag(Wrs, 64, (mt2+0)*16 + col, s, l);
      v16h a1 = load_afrag(Wrs, 64, (mt2+1)*16 + col, s, l);
      #pragma unroll
      for (int nt = 0; nt < 4; ++nt) {
        v16h bm = *(const v16h*)&zfrag[(((nbase+nt)*2 + s)*32 + l)*16];
        rs[0][nt] = wmma_f16(a0, bm, rs[0][nt]);
        rs[1][nt] = wmma_f16(a1, bm, rs[1][nt]);
      }
    }
    #pragma unroll
    for (int m = 0; m < 2; ++m)
      #pragma unroll
      for (int nt = 0; nt < 4; ++nt) {
        int t = t0 + (nbase+nt)*16 + col;
        #pragma unroll
        for (int q = 0; q < 8; ++q) {
          size_t idx = ((size_t)b*RCH + (mt2+m)*16 + rsub + q)*T_FULL + t;
          hout[idx] = (hin[idx] + rs[m][nt][q]) * scale;
        }
      }
  }
}

// ---------------- fc1: relu(skips crop) -> relu(conv K=3) -> y1 (f16) ------
// grid: 161 * 4 * 8 blocks; each block: M=256 slice, N=64 times.
__global__ __launch_bounds__(256) void fc1_kernel(
    const float* __restrict__ skips, const _Float16* __restrict__ Wf1,
    _Float16* __restrict__ y1) {
  __shared__ __align__(32) _Float16 bfrag[384*64];
  int bx = blockIdx.x;
  int tile = bx % 161; int r = bx / 161;
  int b = r & 3; int mblk = r >> 2;
  int t0 = tile << 6;
  int tid = threadIdx.x;

  if (tid < 128) __builtin_prefetch(Wf1 + ((size_t)mblk*256 + tid*2) * 384, 0, 1);

  for (int e = tid; e < 384*64; e += 256) {
    int K = e >> 6, n = e & 63;
    int k = K >> 7, ci = K & 127;
    int t = t0 + n + (k - 1);
    float v = 0.f;
    if (t >= 0 && t < T2) v = skips[((size_t)b*RCH + ci)*T_FULL + LDIFF + t];
    v = v > 0.f ? v : 0.f;                           // relu(skips)
    bfrag[frag_idx(K, n, 12)] = (_Float16)v;
  }
  __syncthreads();

  int w = tid >> 5, l = tid & 31, col = l & 15;
  v8f acc[2][4];
  #pragma unroll
  for (int m = 0; m < 2; ++m)
    #pragma unroll
    for (int nt = 0; nt < 4; ++nt) acc[m][nt] = zero8();

  int mrow = mblk*256 + w*32;
  #pragma unroll 2
  for (int s = 0; s < 12; ++s) {
    v16h a0 = load_afrag(Wf1, 384, mrow + col,      s, l);
    v16h a1 = load_afrag(Wf1, 384, mrow + 16 + col, s, l);
    #pragma unroll
    for (int nt = 0; nt < 4; ++nt) {
      v16h bm = *(const v16h*)&bfrag[((nt*12 + s)*32 + l)*16];
      acc[0][nt] = wmma_f16(a0, bm, acc[0][nt]);
      acc[1][nt] = wmma_f16(a1, bm, acc[1][nt]);
    }
  }

  int rsub = ((l >= 16) ? 8 : 0);
  #pragma unroll
  for (int m = 0; m < 2; ++m)
    #pragma unroll
    for (int nt = 0; nt < 4; ++nt) {
      int t = t0 + nt*16 + col;
      #pragma unroll
      for (int q = 0; q < 8; ++q) {
        int row = mrow + m*16 + q + rsub;
        float o = acc[m][nt][q];
        o = o > 0.f ? o : 0.f;                       // relu(fc1)
        y1[((size_t)b*FC1C + row)*TPAD + t] = (_Float16)o;
      }
    }
}

// ---------------- fc2 (K=6144, 16 LDS chunks) fused with 1x1 last conv -----
// grid: 161 * 4 blocks; block covers M=256 (all fc2 channels), N=64.
__global__ __launch_bounds__(256) void fc2_last_kernel(
    const _Float16* __restrict__ y1, const _Float16* __restrict__ Wf2,
    const float* __restrict__ Wlast, const float* __restrict__ blast,
    float* __restrict__ out) {
  __shared__ __align__(32) _Float16 bfrag[384*64];
  __shared__ float red[64];
  int bx = blockIdx.x;
  int tile = bx % 161; int b = bx / 161;
  int t0 = tile << 6;
  int tid = threadIdx.x;
  int w = tid >> 5, l = tid & 31, col = l & 15;

  v8f acc[2][4];
  #pragma unroll
  for (int m = 0; m < 2; ++m)
    #pragma unroll
    for (int nt = 0; nt < 4; ++nt) acc[m][nt] = zero8();

  for (int ch = 0; ch < 16; ++ch) {
    if (tid < 64) __builtin_prefetch(Wf2 + (size_t)tid*4*6144 + (ch^1)*384, 0, 1);
    for (int e = tid; e < 384*64; e += 256) {
      int Kl = e >> 6, n = e & 63;
      int Kg = ch*384 + Kl;
      int k = Kg / 2048, c = Kg & 2047;
      int t = t0 + n + (k - 1);
      _Float16 v = (_Float16)0.f;
      if (t >= 0 && t < T2) v = y1[((size_t)b*FC1C + c)*TPAD + t];
      bfrag[frag_idx(Kl, n, 12)] = v;
    }
    __syncthreads();
    #pragma unroll 2
    for (int s = 0; s < 12; ++s) {
      v16h a0 = load_afrag(Wf2, 6144, w*32 + col,      ch*12 + s, l);
      v16h a1 = load_afrag(Wf2, 6144, w*32 + 16 + col, ch*12 + s, l);
      #pragma unroll
      for (int nt = 0; nt < 4; ++nt) {
        v16h bm = *(const v16h*)&bfrag[((nt*12 + s)*32 + l)*16];
        acc[0][nt] = wmma_f16(a0, bm, acc[0][nt]);
        acc[1][nt] = wmma_f16(a1, bm, acc[1][nt]);
      }
    }
    __syncthreads();
  }

  // last conv: dot over 256 channels with W_last, reduce via LDS atomics
  if (tid < 64) red[tid] = 0.f;
  __syncthreads();
  int rsub = ((l >= 16) ? 8 : 0);
  #pragma unroll
  for (int m = 0; m < 2; ++m)
    #pragma unroll
    for (int nt = 0; nt < 4; ++nt) {
      float p = 0.f;
      #pragma unroll
      for (int q = 0; q < 8; ++q) {
        int row = w*32 + m*16 + q + rsub;
        p += Wlast[row] * acc[m][nt][q];
      }
      atomicAdd(&red[nt*16 + col], p);
    }
  __syncthreads();
  if (tid < 64) {
    int t = t0 + tid;
    if (t < T2) out[(size_t)b*T2 + t] = red[tid] + blast[0];
  }
}

// ---------------- host launcher --------------------------------------------
extern "C" void kernel_launch(void* const* d_in, const int* in_sizes, int n_in,
                              void* d_out, int out_size, void* d_ws, size_t ws_size,
                              hipStream_t stream) {
  const float* x   = (const float*)d_in[0];
  const float* Wf  = (const float*)d_in[1];
  const float* Wd  = (const float*)d_in[2];
  const float* Ws  = (const float*)d_in[3];
  const float* Wr  = (const float*)d_in[4];
  const float* Wf1 = (const float*)d_in[5];
  const float* Wf2 = (const float*)d_in[6];
  const float* Wl  = (const float*)d_in[7];
  const float* bl  = (const float*)d_in[8];

  char* ws = (char*)d_ws;
  float* hA       = (float*)(ws + OFF_HA);
  float* hB       = (float*)(ws + OFF_HB);
  float* skips    = (float*)(ws + OFF_SKIP);
  _Float16* y1    = (_Float16*)(ws + OFF_Y1);
  _Float16* wd16  = (_Float16*)(ws + OFF_WD);
  _Float16* ws16  = (_Float16*)(ws + OFF_WS);
  _Float16* wr16  = (_Float16*)(ws + OFF_WR);
  _Float16* wf116 = (_Float16*)(ws + OFF_WF1);
  _Float16* wf216 = (_Float16*)(ws + OFF_WF2);

  // weight conversion (idempotent, every call)
  prep_weights<<<16896, 256, 0, stream>>>(Wd, Ws, Wr, Wf1, Wf2,
                                          wd16, ws16, wr16, wf116, wf216);

  const size_t NH = (size_t)4 * RCH * T_FULL;            // 8,388,608
  first_conv<<<(int)((NH + 255) / 256), 256, 0, stream>>>(x, Wf, hA);
  zero_buf<<<(int)((NH + 255) / 256), 256, 0, stream>>>(skips, NH);

  float* hin = hA; float* hout = hB;
  for (int i = 0; i < NLAYERS; ++i) {
    int d = 1 << (i % 10);
    layer_kernel<<<4 * (T_FULL / 128), 256, 0, stream>>>(
        hin, hout, skips,
        wd16 + (size_t)i * 128 * 384,
        ws16 + (size_t)i * 128 * 64,
        wr16 + (size_t)i * 128 * 64, d);
    float* tmp = hin; hin = hout; hout = tmp;
  }

  fc1_kernel<<<161 * 4 * 8, 256, 0, stream>>>(skips, wf116, y1);
  fc2_last_kernel<<<161 * 4, 256, 0, stream>>>(y1, wf216, Wl, bl, (float*)d_out);
}
